// RoIProcessingNetwork_20856361189819
// MI455X (gfx1250) — compile-verified, matrix operand order checked by reading, not run on hardware
//
#include <hip/hip_runtime.h>
#include <hip/hip_bf16.h>
#include <math.h>

// ---------------------------------------------------------------------------
// Types for CDNA5 WMMA (wave32): V_WMMA_F32_16X16X32_BF16
// ---------------------------------------------------------------------------
typedef __attribute__((ext_vector_type(16))) __bf16 v16bf;
typedef __attribute__((ext_vector_type(8)))  __bf16 v8bf;
typedef __attribute__((ext_vector_type(8)))  float  v8f;

#define NCLS   80
#define PP     7
#define SR     2
#define CCH    256
#define IMG    640
#define CLIN   1024
#define C1     (CCH*PP*PP)     // 12544
#define NHEAD  512             // 81 cls + 324 reg, padded to 128-multiple
#define KNMS   500
#define DW_MAX 4.1351665567423560f
#define CLSOFF 4096.0f

// ---------------------------------------------------------------------------
// 1) Multilevel RoIAlign -> bf16 feature matrix [R x 12544]
//    one block per RoI, one thread per channel
// ---------------------------------------------------------------------------
__global__ void roi_align_kernel(const float* __restrict__ rois,
                                 const int*   __restrict__ imidx,
                                 const float* __restrict__ f0,
                                 const float* __restrict__ f1,
                                 const float* __restrict__ f2,
                                 const float* __restrict__ f3,
                                 __bf16* __restrict__ feats)
{
    const int r = blockIdx.x;
    const int c = threadIdx.x;              // channel 0..255

    const float px1 = rois[r*4+0], py1 = rois[r*4+1];
    const float px2 = rois[r*4+2], py2 = rois[r*4+3];

    // level selection
    const float wpx = px2 - px1, hpx = py2 - py1;
    float lv = floorf(4.0f + log2f(sqrtf(fmaxf(wpx*hpx, 1e-6f)) / 224.0f));
    lv = fminf(fmaxf(lv, 2.0f), 5.0f);
    const int l = (int)lv - 2;
    const float stride = (float)(4 << l);
    const int H = IMG >> (2 + l);           // 160,80,40,20 (square maps)
    const float Hf = (float)H;

    const float* fm = (l == 0) ? f0 : (l == 1) ? f1 : (l == 2) ? f2 : f3;
    const size_t base = ((size_t)imidx[r] * CCH + c) * (size_t)H * H;

    const float bx1 = px1 / stride, by1 = py1 / stride;
    const float bx2 = px2 / stride, by2 = py2 / stride;
    const float bw = fmaxf(bx2 - bx1, 1.0f);
    const float bh = fmaxf(by2 - by1, 1.0f);

    for (int py = 0; py < PP; ++py) {
        for (int px = 0; px < PP; ++px) {
            float acc = 0.0f;
            #pragma unroll
            for (int sy = 0; sy < SR; ++sy) {
                const int iy = py*SR + sy;
                float ys = by1 + bh * ((float)iy + 0.5f) / (PP*SR);
                const bool ym = (ys >= -1.0f) && (ys <= Hf);
                ys = fminf(fmaxf(ys, 0.0f), Hf - 1.0f);
                const float y0f = floorf(ys);
                const int   y0  = (int)y0f;
                const int   y1i = min(y0 + 1, H - 1);
                const float ly  = ys - y0f, hy = 1.0f - ly;
                #pragma unroll
                for (int sx = 0; sx < SR; ++sx) {
                    const int ix = px*SR + sx;
                    float xs = bx1 + bw * ((float)ix + 0.5f) / (PP*SR);
                    const bool xm = (xs >= -1.0f) && (xs <= Hf);
                    xs = fminf(fmaxf(xs, 0.0f), Hf - 1.0f);
                    const float x0f = floorf(xs);
                    const int   x0  = (int)x0f;
                    const int   x1i = min(x0 + 1, H - 1);
                    const float lx  = xs - x0f, hx = 1.0f - lx;
                    if (ym && xm) {
                        const float v00 = fm[base + (size_t)y0 * H + x0];
                        const float v01 = fm[base + (size_t)y0 * H + x1i];
                        const float v10 = fm[base + (size_t)y1i * H + x0];
                        const float v11 = fm[base + (size_t)y1i * H + x1i];
                        acc += hy*hx*v00 + hy*lx*v01 + ly*hx*v10 + ly*lx*v11;
                    }
                }
            }
            feats[(size_t)r * C1 + (size_t)c * (PP*PP) + py*PP + px] =
                (__bf16)(acc * (1.0f / (SR*SR)));
        }
    }
}

// ---------------------------------------------------------------------------
// 2) WMMA GEMM:  out[M,N] = act( A[M,K](bf16) * B[K,N](f32->bf16) + bias )
//    block: 256 thr = 8 waves; block tile 32(M) x 128(N); K-step 32.
//    Each wave owns two 16x16 C tiles (one A fragment, two B fragments),
//    issuing 2x V_WMMA_F32_16X16X32_BF16 per K-step.
//    A tile is staged with the CDNA5 async DMA path
//    (global_load_async_to_lds_b128, ASYNCcnt); B is converted f32->bf16 and
//    stored TRANSPOSED ([col][k]) so fragments are contiguous ds_load_b128s.
// ---------------------------------------------------------------------------
__global__ void gemm_wmma_bf16(const __bf16* __restrict__ A,
                               const float*  __restrict__ B,
                               const float*  __restrict__ bias,
                               float*        outF,      // may be null
                               __bf16*       outB,      // may be null
                               int M, int N, int K, int relu)
{
    __shared__ __bf16 lA [32 * 32];    // [row][k]  row-major, 2 KB
    __shared__ __bf16 lBT[128 * 32];   // [col][k]  transposed, 8 KB

    const int t     = threadIdx.x;
    const int ntile = blockIdx.x;
    const int mtile = blockIdx.y;
    const int w     = t >> 5;
    const int lane  = t & 31;
    const int wm    = w >> 2;       // 0..1  (16-row subtile)
    const int wn    = w & 3;        // 0..3  (32-col group)
    const int half  = lane >> 4;    // 0/1
    const int l16   = lane & 15;

    const int row0 = mtile * 32;
    const int col0 = ntile * 128;

    v8f acc0 = {0.f,0.f,0.f,0.f,0.f,0.f,0.f,0.f};
    v8f acc1 = {0.f,0.f,0.f,0.f,0.f,0.f,0.f,0.f};

    // staging assignments
    const int acp_r  = t >> 2;          // A copy: row (t<128)
    const int acp_ch = t & 3;           // 4 chunks of 8 bf16 per 32-wide row
    const int bcg    = t & 15;          // B: 16 groups of 8 cols
    const int bkp    = t >> 4;          // B: 16 k-pairs (2 rows each)

    const unsigned lA_dst =
        (unsigned)(uintptr_t)&lA[acp_r * 32 + acp_ch * 8];

    for (int k0 = 0; k0 < K; k0 += 32) {
        // ---- stage A tile (32 rows x 32 k bf16) via async DMA to LDS ----
        if (t < 128) {
            const unsigned long long ga = (unsigned long long)(uintptr_t)
                (A + (size_t)(row0 + acp_r) * K + k0 + acp_ch * 8);
            asm volatile("global_load_async_to_lds_b128 %0, %1, off"
                         :: "v"(lA_dst), "v"(ga) : "memory");
        }
        // ---- stage B tile (32 k x 128 n) transposed, f32 -> bf16 packed ----
        {
            const int kk = k0 + bkp * 2;
            const int cc = bcg * 8;
            const float4 r0a = *(const float4*)(B + (size_t)kk     * N + col0 + cc);
            const float4 r0b = *(const float4*)(B + (size_t)kk     * N + col0 + cc + 4);
            const float4 r1a = *(const float4*)(B + (size_t)(kk+1) * N + col0 + cc);
            const float4 r1b = *(const float4*)(B + (size_t)(kk+1) * N + col0 + cc + 4);
            union { unsigned u; __bf16 h[2]; } pk;
            const float a0[8] = {r0a.x,r0a.y,r0a.z,r0a.w, r0b.x,r0b.y,r0b.z,r0b.w};
            const float a1[8] = {r1a.x,r1a.y,r1a.z,r1a.w, r1b.x,r1b.y,r1b.z,r1b.w};
            #pragma unroll
            for (int j = 0; j < 8; ++j) {
                pk.h[0] = (__bf16)a0[j];
                pk.h[1] = (__bf16)a1[j];
                *(unsigned*)&lBT[(cc + j) * 32 + bkp * 2] = pk.u;
            }
            // prefetch next B tile rows (global_prefetch_b8)
            if (k0 + 32 < K)
                __builtin_prefetch(B + (size_t)(kk + 32) * N + col0 + cc, 0, 1);
        }
        // complete async A-tile DMA before publishing the tile
        asm volatile("s_wait_asynccnt 0x0" ::: "memory");
        __syncthreads();

        // ---- fragments per ISA 16-bit layout ----
        // lanes 0-15: K {0..7, 16..23}; lanes 16-31: K {8..15, 24..31}
        const int kof  = half * 8;
        const int arow = wm * 16 + l16;
        const v8bf a_lo = *(const v8bf*)&lA[arow * 32 + kof];
        const v8bf a_hi = *(const v8bf*)&lA[arow * 32 + kof + 16];
        const v16bf af = __builtin_shufflevector(a_lo, a_hi,
                            0,1,2,3,4,5,6,7,8,9,10,11,12,13,14,15);

        const int bc0 = wn * 32 + l16;
        const v8bf b0_lo = *(const v8bf*)&lBT[bc0 * 32 + kof];
        const v8bf b0_hi = *(const v8bf*)&lBT[bc0 * 32 + kof + 16];
        const v16bf bf0 = __builtin_shufflevector(b0_lo, b0_hi,
                            0,1,2,3,4,5,6,7,8,9,10,11,12,13,14,15);

        const int bc1 = bc0 + 16;
        const v8bf b1_lo = *(const v8bf*)&lBT[bc1 * 32 + kof];
        const v8bf b1_hi = *(const v8bf*)&lBT[bc1 * 32 + kof + 16];
        const v16bf bf1 = __builtin_shufflevector(b1_lo, b1_hi,
                            0,1,2,3,4,5,6,7,8,9,10,11,12,13,14,15);

        acc0 = __builtin_amdgcn_wmma_f32_16x16x32_bf16(
                   false, af, false, bf0, (short)0, acc0, false, false);
        acc1 = __builtin_amdgcn_wmma_f32_16x16x32_bf16(
                   false, af, false, bf1, (short)0, acc1, false, false);
        __syncthreads();
    }

    // ---- epilogue: bias + activation + store (two 16-col subtiles) ----
    #pragma unroll
    for (int sub = 0; sub < 2; ++sub) {
        const v8f& acc = sub ? acc1 : acc0;
        const int ocol = col0 + wn * 32 + sub * 16 + l16;
        const float bv = bias ? bias[ocol] : 0.0f;
        #pragma unroll
        for (int r = 0; r < 8; ++r) {
            const int orow = row0 + wm * 16 + half * 8 + r;
            float v = acc[r] + bv;
            if (relu) v = v > 0.0f ? v : 0.0f;
            if (outF) outF[(size_t)orow * N + ocol] = v;
            if (outB) outB[(size_t)orow * N + ocol] = (__bf16)v;
        }
    }
}

// ---------------------------------------------------------------------------
// 3) Pack cls/reg head weights into one padded [1024 x 512] f32 matrix
// ---------------------------------------------------------------------------
__global__ void pad_head_w(const float* __restrict__ Wc,
                           const float* __restrict__ Wr,
                           float* __restrict__ Wh)
{
    const int t = blockIdx.x * blockDim.x + threadIdx.x;
    if (t >= CLIN * NHEAD) return;
    const int k = t / NHEAD, j = t % NHEAD;
    float v = 0.0f;
    if (j < 81)       v = Wc[k * 81 + j];
    else if (j < 405) v = Wr[k * 324 + (j - 81)];
    Wh[t] = v;
}
__global__ void pad_head_b(const float* __restrict__ bc,
                           const float* __restrict__ br,
                           float* __restrict__ bh)
{
    const int t = blockIdx.x * blockDim.x + threadIdx.x;
    if (t >= NHEAD) return;
    bh[t] = (t < 81) ? bc[t] : ((t < 405) ? br[t - 81] : 0.0f);
}

// ---------------------------------------------------------------------------
// 4) softmax stats per RoI over the 81 class logits
// ---------------------------------------------------------------------------
__global__ void softmax_stats(const float* __restrict__ head,
                              float* __restrict__ rs, int R)
{
    const int r = blockIdx.x * blockDim.x + threadIdx.x;
    if (r >= R) return;
    const float* p = head + (size_t)r * NHEAD;
    float m = p[0];
    for (int c = 1; c < 81; ++c) m = fmaxf(m, p[c]);
    float sum = 0.0f;
    for (int c = 0; c < 81; ++c) sum += expf(p[c] - m);
    rs[r * 2 + 0] = m;
    rs[r * 2 + 1] = sum;
}

// ---------------------------------------------------------------------------
// 5) Box decode + validity per (roi, class)
// ---------------------------------------------------------------------------
__global__ void decode_kernel(const float* __restrict__ head,
                              const float* __restrict__ rs,
                              const float* __restrict__ rois,
                              const int*   __restrict__ imidx,
                              const int*   __restrict__ imsizes,
                              const float* __restrict__ score_thr_p,
                              const float* __restrict__ min_size_p,
                              float* __restrict__ boxes,
                              float* __restrict__ scr,
                              int*   __restrict__ valid,
                              int R)
{
    const int idx = blockIdx.x * blockDim.x + threadIdx.x;
    if (idx >= R * NCLS) return;
    const int r = idx / NCLS, c = idx % NCLS;
    const int cls = c + 1;
    const float* hp = head + (size_t)r * NHEAD;

    const float s = expf(hp[cls] - rs[r*2]) / rs[r*2+1];

    const float rx = hp[81 + cls*4 + 0];
    const float ry = hp[81 + cls*4 + 1];
    const float rw = hp[81 + cls*4 + 2];
    const float rh = hp[81 + cls*4 + 3];

    const float px1 = rois[r*4+0], py1 = rois[r*4+1];
    const float px2 = rois[r*4+2], py2 = rois[r*4+3];
    const float cx = (px1 + px2) * 0.5f, cy = (py1 + py2) * 0.5f;
    const float pw = px2 - px1, ph = py2 - py1;

    const float dx = rx * 0.1f, dy = ry * 0.1f;
    const float dw = fminf(rw * 0.2f, DW_MAX);
    const float dh = fminf(rh * 0.2f, DW_MAX);

    const float pcx = dx * pw + cx, pcy = dy * ph + cy;
    const float ppw = expf(dw) * pw, pph = expf(dh) * ph;

    const int im = imidx[r];
    const float Wd = (float)imsizes[im*2+1];
    const float Hd = (float)imsizes[im*2+0];

    const float bx1 = fminf(fmaxf(pcx - 0.5f*ppw, 0.0f), Wd);
    const float by1 = fminf(fmaxf(pcy - 0.5f*pph, 0.0f), Hd);
    const float bx2 = fminf(fmaxf(pcx + 0.5f*ppw, 0.0f), Wd);
    const float by2 = fminf(fmaxf(pcy + 0.5f*pph, 0.0f), Hd);

    boxes[idx*4+0] = bx1; boxes[idx*4+1] = by1;
    boxes[idx*4+2] = bx2; boxes[idx*4+3] = by2;
    scr[idx] = s;
    const bool v = (s > *score_thr_p) &&
                   (bx2 - bx1 >= *min_size_p) &&
                   (by2 - by1 >= *min_size_p);
    valid[idx] = v ? 1 : 0;
}

// ---------------------------------------------------------------------------
// 6) Per-image: top-500 -> class-offset NMS -> top-imtop writeout
//    one block (512 thr) per image; tie-break matches lax.top_k (index order)
// ---------------------------------------------------------------------------
__global__ void postproc_kernel(const float* __restrict__ boxes,
                                const float* __restrict__ scr,
                                const int*   __restrict__ valid,
                                const int*   __restrict__ imidx,
                                const float* __restrict__ iou_thr_p,
                                float* __restrict__ simg,
                                float* __restrict__ out,
                                int R, int n, int imtop)
{
    const int i = blockIdx.x;
    const int t = threadIdx.x;            // 512 threads
    const int TOT = R * NCLS;
    float* s = simg + (size_t)i * TOT;

    for (int j = t; j < TOT; j += 512)
        s[j] = (valid[j] && imidx[j / NCLS] == i) ? scr[j] : -1.0f;
    __syncthreads();

    __shared__ float rv[512];
    __shared__ int   ri[512];
    __shared__ float ltv[KNMS];
    __shared__ int   lti[KNMS];
    __shared__ float lvb[KNMS*4];
    __shared__ int   lvc[KNMS];
    __shared__ int   lvm[KNMS];
    __shared__ float lob[KNMS*4];
    __shared__ int   keep[KNMS];
    __shared__ int   supp[KNMS];
    __shared__ float ls2[KNMS];

    // ---- top KNMS via repeated block argmax ----
    for (int k = 0; k < KNMS; ++k) {
        float bv = -1e38f; int bi = 0x7fffffff;
        for (int j = t; j < TOT; j += 512) {
            const float v = s[j];
            if (v > bv) { bv = v; bi = j; }
        }
        rv[t] = bv; ri[t] = bi;
        __syncthreads();
        for (int sh = 256; sh > 0; sh >>= 1) {
            if (t < sh) {
                if (rv[t+sh] > rv[t] || (rv[t+sh] == rv[t] && ri[t+sh] < ri[t])) {
                    rv[t] = rv[t+sh]; ri[t] = ri[t+sh];
                }
            }
            __syncthreads();
        }
        if (t == 0) { ltv[k] = rv[0]; lti[k] = ri[0]; s[ri[0]] = -1e30f; }
        __syncthreads();
    }

    // ---- gather candidates ----
    if (t < KNMS) {
        const int j = lti[t];
        const int cls = (j % NCLS) + 1;
        lvb[t*4+0] = boxes[j*4+0]; lvb[t*4+1] = boxes[j*4+1];
        lvb[t*4+2] = boxes[j*4+2]; lvb[t*4+3] = boxes[j*4+3];
        lvc[t] = cls;
        lvm[t] = (valid[j] && imidx[j / NCLS] == i) ? 1 : 0;
        const float off = (float)cls * CLSOFF;
        lob[t*4+0] = lvb[t*4+0] + off; lob[t*4+1] = lvb[t*4+1] + off;
        lob[t*4+2] = lvb[t*4+2] + off; lob[t*4+3] = lvb[t*4+3] + off;
        keep[t] = 0; supp[t] = 0;
    }
    __syncthreads();

    // ---- sequential NMS (parallel over j) ----
    const float thr = *iou_thr_p;
    for (int ii = 0; ii < KNMS; ++ii) {
        const int ki = lvm[ii] && !supp[ii];
        if (t == 0) keep[ii] = ki;
        if (ki && t < KNMS && t > ii) {
            const float ax1 = lob[ii*4+0], ay1 = lob[ii*4+1];
            const float ax2 = lob[ii*4+2], ay2 = lob[ii*4+3];
            const float bx1 = lob[t*4+0],  by1 = lob[t*4+1];
            const float bx2 = lob[t*4+2],  by2 = lob[t*4+3];
            const float aa = fmaxf(ax2-ax1,0.f) * fmaxf(ay2-ay1,0.f);
            const float ba = fmaxf(bx2-bx1,0.f) * fmaxf(by2-by1,0.f);
            const float ix1 = fmaxf(ax1,bx1), iy1 = fmaxf(ay1,by1);
            const float ix2 = fminf(ax2,bx2), iy2 = fminf(ay2,by2);
            const float inter = fmaxf(ix2-ix1,0.f) * fmaxf(iy2-iy1,0.f);
            const float iou = inter / fmaxf(aa + ba - inter, 1e-6f);
            if (iou > thr) supp[t] = 1;
        }
        __syncthreads();
    }

    if (t < KNMS) ls2[t] = keep[t] ? ltv[t] : -1.0f;
    __syncthreads();

    // ---- final top imtop, writeout ----
    float* bo = out;
    float* so = out + (size_t)n * imtop * 4;
    float* co = out + (size_t)n * imtop * 5;
    for (int k = 0; k < imtop; ++k) {
        rv[t] = (t < KNMS) ? ls2[t] : -1e38f;
        ri[t] = (t < KNMS) ? t : 0x7fffffff;
        __syncthreads();
        for (int sh = 256; sh > 0; sh >>= 1) {
            if (t < sh) {
                if (rv[t+sh] > rv[t] || (rv[t+sh] == rv[t] && ri[t+sh] < ri[t])) {
                    rv[t] = rv[t+sh]; ri[t] = ri[t+sh];
                }
            }
            __syncthreads();
        }
        const int fi = ri[0];
        if (t == 0) {
            const int km = keep[fi];
            bo[((size_t)i*imtop + k)*4 + 0] = km ? lvb[fi*4+0] : 0.0f;
            bo[((size_t)i*imtop + k)*4 + 1] = km ? lvb[fi*4+1] : 0.0f;
            bo[((size_t)i*imtop + k)*4 + 2] = km ? lvb[fi*4+2] : 0.0f;
            bo[((size_t)i*imtop + k)*4 + 3] = km ? lvb[fi*4+3] : 0.0f;
            so[(size_t)i*imtop + k] = km ? ltv[fi] : 0.0f;
            co[(size_t)i*imtop + k] = km ? (float)lvc[fi] : -1.0f;
            ls2[fi] = -1e30f;
        }
        __syncthreads();
    }
}

// ---------------------------------------------------------------------------
// Host launcher
// ---------------------------------------------------------------------------
static inline char* ws_carve(char*& p, size_t bytes) {
    char* ret = p;
    size_t b = (bytes + 255) & ~(size_t)255;
    p += b;
    return ret;
}

extern "C" void kernel_launch(void* const* d_in, const int* in_sizes, int n_in,
                              void* d_out, int out_size, void* d_ws, size_t ws_size,
                              hipStream_t stream) {
    const float* proposals = (const float*)d_in[0];
    const int*   imidx     = (const int*)  d_in[1];
    const float* f0        = (const float*)d_in[2];
    const float* f1        = (const float*)d_in[3];
    const float* f2        = (const float*)d_in[4];
    const float* f3        = (const float*)d_in[5];
    const int*   imsizes   = (const int*)  d_in[6];
    const float* W1        = (const float*)d_in[7];
    const float* b1        = (const float*)d_in[8];
    const float* W2        = (const float*)d_in[9];
    const float* b2        = (const float*)d_in[10];
    const float* Wc        = (const float*)d_in[11];
    const float* bc        = (const float*)d_in[12];
    const float* Wr        = (const float*)d_in[13];
    const float* br        = (const float*)d_in[14];
    const float* score_thr = (const float*)d_in[15];
    const float* iou_thr   = (const float*)d_in[16];
    const float* min_size  = (const float*)d_in[17];
    (void)d_in[18]; (void)n_in; (void)ws_size;

    const int R = in_sizes[0] / 4;         // 512
    const int n = in_sizes[6] / 2;         // 2
    const int imtop = out_size / (6 * n);  // 100

    char* p = (char*)d_ws;
    __bf16* feats = (__bf16*)ws_carve(p, (size_t)R * C1 * sizeof(__bf16));
    __bf16* x1b   = (__bf16*)ws_carve(p, (size_t)R * CLIN * sizeof(__bf16));
    __bf16* x2b   = (__bf16*)ws_carve(p, (size_t)R * CLIN * sizeof(__bf16));
    float*  Whf   = (float*) ws_carve(p, (size_t)CLIN * NHEAD * sizeof(float));
    float*  bhf   = (float*) ws_carve(p, (size_t)NHEAD * sizeof(float));
    float*  head  = (float*) ws_carve(p, (size_t)R * NHEAD * sizeof(float));
    float*  rs    = (float*) ws_carve(p, (size_t)R * 2 * sizeof(float));
    float*  boxes = (float*) ws_carve(p, (size_t)R * NCLS * 4 * sizeof(float));
    float*  scr   = (float*) ws_carve(p, (size_t)R * NCLS * sizeof(float));
    int*    valid = (int*)   ws_carve(p, (size_t)R * NCLS * sizeof(int));
    float*  simg  = (float*) ws_carve(p, (size_t)n * R * NCLS * sizeof(float));

    // 1) RoIAlign -> bf16 features
    roi_align_kernel<<<R, CCH, 0, stream>>>(proposals, imidx, f0, f1, f2, f3, feats);

    // 2) FC1: [R x 12544] * [12544 x 1024] + b1, ReLU -> bf16
    gemm_wmma_bf16<<<dim3(CLIN/128, R/32), 256, 0, stream>>>(
        feats, W1, b1, nullptr, x1b, R, CLIN, C1, 1);

    // 3) FC2: [R x 1024] * [1024 x 1024] + b2, ReLU -> bf16
    gemm_wmma_bf16<<<dim3(CLIN/128, R/32), 256, 0, stream>>>(
        x1b, W2, b2, nullptr, x2b, R, CLIN, CLIN, 1);

    // 4) pack heads, then combined head GEMM -> f32 [R x 512]
    pad_head_w<<<(CLIN*NHEAD + 255)/256, 256, 0, stream>>>(Wc, Wr, Whf);
    pad_head_b<<<(NHEAD + 255)/256, 256, 0, stream>>>(bc, br, bhf);
    gemm_wmma_bf16<<<dim3(NHEAD/128, R/32), 256, 0, stream>>>(
        x2b, Whf, bhf, head, nullptr, R, NHEAD, CLIN, 0);

    // 5) softmax stats + decode
    softmax_stats<<<(R + 255)/256, 256, 0, stream>>>(head, rs, R);
    decode_kernel<<<(R*NCLS + 255)/256, 256, 0, stream>>>(
        head, rs, proposals, imidx, imsizes, score_thr, min_size,
        boxes, scr, valid, R);

    // 6) per-image top-k / NMS / writeout
    postproc_kernel<<<n, 512, 0, stream>>>(
        boxes, scr, valid, imidx, iou_thr, simg, (float*)d_out, R, n, imtop);
}
